// WLN_Edit_970662609324
// MI455X (gfx1250) — matrix operand order repaired and checked by reading, not compiled, not run on hardware
//
#include <hip/hip_runtime.h>

// ---------------------------------------------------------------------------
// WLN message passing on MI455X (gfx1250, wave32).
// - All GEMMs: V_WMMA_F32_16X16X32_BF16, f32 accumulate.
// - atom_features / nei_label live in workspace as packed-bf16 pairs
//   ([row][128] dwords) == exactly the A-fragment LDS layout, so tile staging
//   and the U2 neighbor gather are raw row copies done with CDNA5 async DMA:
//   GLOBAL_LOAD_ASYNC_TO_LDS_B128 (+ s_wait_asynccnt), no VGPR round-trip.
// - Weights repacked once into fragment-major bf16 pairs (two b128 per
//   B-fragment per lane).
// - U2 fuses gather -> GEMM -> relu(+bias) -> masked 10:1 neighbor sum.
// ---------------------------------------------------------------------------

typedef __attribute__((ext_vector_type(16))) __bf16 v16bf;
typedef __attribute__((ext_vector_type(8)))  float  v8f;

#define B_    64
#define N_    512
#define NB_   600
#define A_    89
#define E_    5
#define H_    256
#define MAXNB 10

// round-to-nearest-even f32 -> bf16 pair (lo = a, hi = b)
static __device__ __forceinline__ unsigned pack_bf16(float a, float b) {
  unsigned ua = __float_as_uint(a), ub = __float_as_uint(b);
  ua += 0x7FFFu + ((ua >> 16) & 1u);
  ub += 0x7FFFu + ((ub >> 16) & 1u);
  return (ua >> 16) | (ub & 0xFFFF0000u);
}
static __device__ __forceinline__ unsigned short bf16_1(float a) {
  unsigned ua = __float_as_uint(a);
  ua += 0x7FFFu + ((ua >> 16) & 1u);
  return (unsigned short)(ua >> 16);
}

// bf16 fragment from two 16B chunks (lowers to 2x ds_load_b128 / global_load_b128)
static __device__ __forceinline__ v16bf frag2(const unsigned* p0, const unsigned* p1) {
  union { uint4 q[2]; v16bf bf; } u;
  u.q[0] = *reinterpret_cast<const uint4*>(p0);
  u.q[1] = *reinterpret_cast<const uint4*>(p1);
  return u.bf;
}

// generic->LDS byte offset (low 32 bits of the shared-aperture flat address)
static __device__ __forceinline__ unsigned lds_addr(const void* p) {
  return (unsigned)(uintptr_t)p;
}
// CDNA5 async DMA: 16B global -> LDS, tracked by ASYNCcnt (GV addressing)
static __device__ __forceinline__ void async_b128(unsigned lds, const unsigned* g) {
  asm volatile("global_load_async_to_lds_b128 %0, %1, off"
               :: "v"(lds), "v"(g) : "memory");
}
static __device__ __forceinline__ void wait_async0() {
  asm volatile("s_wait_asynccnt 0x0" ::: "memory");
}

// ---------------------------------------------------------------------------
// Pack weights W[K][256] (f32) -> fragment-major bf16 pairs:
//   Wf[(ks*256 + n)*16 + j] = {W[2*(ks*16+j)][n], W[2*(ks*16+j)+1][n]}
// Lane (col n, half h) reads its 32-K B fragment as 8 consecutive dwords.
// ---------------------------------------------------------------------------
__global__ void pack_w_kernel(const float* __restrict__ W, int K, int total,
                              unsigned* __restrict__ Wf) {
  int idx = blockIdx.x * blockDim.x + threadIdx.x;
  if (idx >= total) return;
  int j  = idx & 15;
  int n  = (idx >> 4) & 255;
  int ks = idx >> 12;
  int k0 = (ks * 16 + j) * 2;
  float a = (k0     < K) ? W[k0 * H_ + n]       : 0.f;
  float b = (k0 + 1 < K) ? W[(k0 + 1) * H_ + n] : 0.f;
  Wf[idx] = pack_bf16(a, b);
}

// ---------------------------------------------------------------------------
// Embedding GEMM: Fout(bf16) = input_atom(f32) @ W_atom   (K 89 -> pad 96)
// ---------------------------------------------------------------------------
__global__ __launch_bounds__(128)
void embed_gemm_kernel(const float* __restrict__ src,
                       const unsigned* __restrict__ Wf,
                       unsigned short* __restrict__ out) {
  constexpr int K0 = A_, KK = 48, LROW = 52, KS = 3;
  extern __shared__ unsigned smem[];
  const int tid   = threadIdx.x;
  const int mbase = blockIdx.x * 64;
  const int cbase = blockIdx.y * 64;

  for (int i = tid; i < 64 * KK; i += 128) {
    int row = i / KK, kk = i - row * KK;
    int k = kk * 2, grow = mbase + row;
    float a = (k     < K0) ? src[grow * K0 + k]     : 0.f;
    float b = (k + 1 < K0) ? src[grow * K0 + k + 1] : 0.f;
    smem[row * LROW + kk] = pack_bf16(a, b);
  }
  __syncthreads();

  const int wave = tid >> 5, lane = tid & 31;
  const int ln = lane & 15, h = lane >> 4;
  const int col = cbase + wave * 16 + ln;

  v8f acc[4] = {};
  for (int ks = 0; ks < KS; ++ks) {
    const unsigned* bp = Wf + ((ks * 256 + col) * 16 + h * 8);
    v16bf bfrag = frag2(bp, bp + 4);
#pragma unroll
    for (int t = 0; t < 4; ++t) {
      const unsigned* ap = smem + ((16 * t + ln) * LROW + ks * 16 + h * 8);
      acc[t] = __builtin_amdgcn_wmma_f32_16x16x32_bf16(
          false, frag2(ap, ap + 4), false, bfrag, (short)0, acc[t], false, false);
    }
  }
#pragma unroll
  for (int t = 0; t < 4; ++t)
#pragma unroll
    for (int v = 0; v < 8; ++v)
      out[(mbase + 16 * t + v + 8 * h) * H_ + col] = bf16_1(acc[t][v]);
}

// ---------------------------------------------------------------------------
// U1 GEMM: out = relu(concat(F, nei) @ W_U1 + b_U1), K = 512.
// A tile staged with async DMA from the two packed-bf16 sources.
// ---------------------------------------------------------------------------
template <bool OUT_F32>
__global__ __launch_bounds__(128)
void u1_gemm_kernel(const unsigned* __restrict__ Fp,
                    const unsigned* __restrict__ NEIp,
                    const unsigned* __restrict__ Wf,
                    const float* __restrict__ bias,
                    void* __restrict__ outv) {
  constexpr int LROW = 260, KS = 16;  // 256 data dwords + 4 skew
  extern __shared__ unsigned smem[];
  const int tid   = threadIdx.x;
  const int mbase = blockIdx.x * 64;
  const int cbase = blockIdx.y * 64;
  const int wave = tid >> 5, lane = tid & 31;
  const int ln = lane & 15, h = lane >> 4;

  // stage A: per row, 512B from F + 512B from NEI, 16B per lane, async DMA
  for (int r = wave; r < 64; r += 4) {
    const unsigned* g0 = Fp   + (size_t)(mbase + r) * 128 + lane * 4;
    const unsigned* g1 = NEIp + (size_t)(mbase + r) * 128 + lane * 4;
    unsigned l0 = lds_addr(smem + r * LROW + lane * 4);
    async_b128(l0, g0);
    async_b128(l0 + 512, g1);
  }
  wait_async0();
  __syncthreads();

  const int col = cbase + wave * 16 + ln;
  v8f acc[4] = {};
  for (int ks = 0; ks < KS; ++ks) {
    const unsigned* bp = Wf + ((ks * 256 + col) * 16 + h * 8);
    v16bf bfrag = frag2(bp, bp + 4);
#pragma unroll
    for (int t = 0; t < 4; ++t) {
      const unsigned* ap = smem + ((16 * t + ln) * LROW + ks * 16 + h * 8);
      acc[t] = __builtin_amdgcn_wmma_f32_16x16x32_bf16(
          false, frag2(ap, ap + 4), false, bfrag, (short)0, acc[t], false, false);
    }
  }

  float bv = bias[col];
#pragma unroll
  for (int t = 0; t < 4; ++t) {
#pragma unroll
    for (int v = 0; v < 8; ++v) {
      int row = mbase + 16 * t + v + 8 * h;
      float val = fmaxf(acc[t][v] + bv, 0.f);
      if (OUT_F32) ((float*)outv)[(size_t)row * H_ + col] = val;
      else         ((unsigned short*)outv)[(size_t)row * H_ + col] = bf16_1(val);
    }
  }
}

// ---------------------------------------------------------------------------
// Fused U2: 16 atoms x 10 neighbors per block.
//   async-DMA gather of bf16 feature rows -> LDS, bond tail packed inline,
//   10x WMMA tiles x 9 K-steps, relu(+bias)*mask, 10:1 sum -> nei (bf16).
// Reduction scratch aliases the dead A tile.
// ---------------------------------------------------------------------------
__global__ __launch_bounds__(128)
void u2_gather_gemm_kernel(const unsigned* __restrict__ Fp,
                           const float* __restrict__ input_bond,
                           const int* __restrict__ atom_graph,
                           const int* __restrict__ bond_graph,
                           const int* __restrict__ num_nbs,
                           const unsigned* __restrict__ WfU2,
                           const float* __restrict__ bU2,
                           unsigned* __restrict__ neip) {
  constexpr int KK = 144, LROW = 148, KS = 9, RST = 66;
  extern __shared__ unsigned smem[];
  int*   nnb = (int*)(smem + 160 * LROW);  // 16 dwords, above A/red regions
  float* red = (float*)smem;               // aliases A tile after WMMA

  const int tid   = threadIdx.x;
  const int abase = blockIdx.x * 16;
  const int cbase = blockIdx.y * 64;
  const int wave = tid >> 5, lane = tid & 31;
  const int ln = lane & 15, h = lane >> 4;

  if (tid < 16) nnb[tid] = num_nbs[abase + tid];

  // gather 160 feature rows (512B each) with async DMA: 1 instr per row/wave
  for (int r = wave; r < 160; r += 4) {
    int a = r / 10, nb = r - a * 10;
    int g = abase + a, b = g >> 9;  // / N_
    int ai = atom_graph[(g * MAXNB + nb) * 2 + 1];
    async_b128(lds_addr(smem + r * LROW + lane * 4),
               Fp + (size_t)(b * N_ + ai) * 128 + lane * 4);
  }
  // bond tail: k = 256..260 then zeros to 287  (disjoint LDS dwords)
  for (int r = tid; r < 160; r += 128) {
    int a = r / 10, nb = r - a * 10;
    int g = abase + a, b = g >> 9;
    int bi = bond_graph[(g * MAXNB + nb) * 2 + 1];
    const float* bp = input_bond + (size_t)(b * NB_ + bi) * E_;
    unsigned* lrow = smem + r * LROW;
    lrow[128] = pack_bf16(bp[0], bp[1]);
    lrow[129] = pack_bf16(bp[2], bp[3]);
    lrow[130] = pack_bf16(bp[4], 0.f);
    for (int kk = 131; kk < KK; ++kk) lrow[kk] = 0u;
  }
  wait_async0();
  __syncthreads();

  const int col = cbase + wave * 16 + ln;
  v8f acc[10] = {};
  for (int ks = 0; ks < KS; ++ks) {
    const unsigned* bp = WfU2 + ((ks * 256 + col) * 16 + h * 8);
    v16bf bfrag = frag2(bp, bp + 4);
#pragma unroll
    for (int t = 0; t < 10; ++t) {
      const unsigned* ap = smem + ((16 * t + ln) * LROW + ks * 16 + h * 8);
      acc[t] = __builtin_amdgcn_wmma_f32_16x16x32_bf16(
          false, frag2(ap, ap + 4), false, bfrag, (short)0, acc[t], false, false);
    }
  }

  float bv = bU2[col];
  __syncthreads();  // A tile dead; reuse as reduction scratch
  const int lc = wave * 16 + ln;
#pragma unroll
  for (int t = 0; t < 10; ++t) {
#pragma unroll
    for (int v = 0; v < 8; ++v) {
      int row = 16 * t + v + 8 * h;
      int a = row / 10, nb = row - a * 10;
      float val = fmaxf(acc[t][v] + bv, 0.f);
      red[row * RST + lc] = (nb < nnb[a]) ? val : 0.f;
    }
  }
  __syncthreads();

  // 10:1 neighbor sum -> packed bf16 nei tile
  for (int i = tid; i < 16 * 32; i += 128) {
    int a = i >> 5, c2 = i & 31;
    const float* rp = red + (a * 10) * RST + c2 * 2;
    float s0 = 0.f, s1 = 0.f;
#pragma unroll
    for (int nb = 0; nb < 10; ++nb) { s0 += rp[nb * RST]; s1 += rp[nb * RST + 1]; }
    neip[(size_t)(abase + a) * 128 + (cbase >> 1) + c2] = pack_bf16(s0, s1);
  }
}

// ---------------------------------------------------------------------------
extern "C" void kernel_launch(void* const* d_in, const int* in_sizes, int n_in,
                              void* d_out, int out_size, void* d_ws, size_t ws_size,
                              hipStream_t stream) {
  (void)in_sizes; (void)n_in; (void)out_size; (void)ws_size;
  const float* input_atom = (const float*)d_in[0];
  const float* input_bond = (const float*)d_in[1];
  const int*   atom_graph = (const int*)d_in[2];
  const int*   bond_graph = (const int*)d_in[3];
  const int*   num_nbs    = (const int*)d_in[4];
  const float* W_atom     = (const float*)d_in[5];
  const float* W_U2       = (const float*)d_in[6];
  const float* b_U2       = (const float*)d_in[7];
  const float* W_U1       = (const float*)d_in[8];
  const float* b_U1       = (const float*)d_in[9];
  float* out = (float*)d_out;

  const size_t M = (size_t)B_ * N_;  // 32768
  char* ws = (char*)d_ws;
  unsigned* F0   = (unsigned*)ws; ws += M * 128 * 4;          // bf16-packed feats
  unsigned* F1   = (unsigned*)ws; ws += M * 128 * 4;
  unsigned* NEI  = (unsigned*)ws; ws += M * 128 * 4;          // bf16-packed nei
  unsigned* WfA  = (unsigned*)ws; ws += (size_t)3  * 256 * 16 * 4;
  unsigned* WfU2 = (unsigned*)ws; ws += (size_t)9  * 256 * 16 * 4;
  unsigned* WfU1 = (unsigned*)ws; ws += (size_t)16 * 256 * 16 * 4;

  {
    int tA = 3 * 256 * 16;
    pack_w_kernel<<<(tA + 255) / 256, 256, 0, stream>>>(W_atom, A_, tA, WfA);
    int t2 = 9 * 256 * 16;
    pack_w_kernel<<<(t2 + 255) / 256, 256, 0, stream>>>(W_U2, H_ + E_, t2, WfU2);
    int t1 = 16 * 256 * 16;
    pack_w_kernel<<<(t1 + 255) / 256, 256, 0, stream>>>(W_U1, 2 * H_, t1, WfU1);
  }

  dim3 blk(128);
  dim3 gg((unsigned)(M / 64), 4);
  embed_gemm_kernel<<<gg, blk, 64 * 52 * 4, stream>>>(
      input_atom, WfA, (unsigned short*)F0);

  dim3 g2((unsigned)(M / 16), 4);
  size_t lds2 = (size_t)(160 * 148 + 16) * 4;  // ~93 KB
  size_t lds1 = (size_t)64 * 260 * 4;          // ~65 KB

  const unsigned* Fin = F0;
  for (int d = 0; d < 3; ++d) {
    u2_gather_gemm_kernel<<<g2, blk, lds2, stream>>>(
        Fin, input_bond, atom_graph, bond_graph, num_nbs, WfU2, b_U2, NEI);
    if (d < 2) {
      unsigned* Fout = (d == 0) ? F1 : F0;
      u1_gemm_kernel<false><<<gg, blk, lds1, stream>>>(Fin, NEI, WfU1, b_U1, Fout);
      Fin = Fout;
    } else {
      u1_gemm_kernel<true><<<gg, blk, lds1, stream>>>(Fin, NEI, WfU1, b_U1, out);
    }
  }
}
